// ST_GCN2D_8710193677082
// MI455X (gfx1250) — matrix operand order repaired and checked by reading, not compiled, not run on hardware
//
#include <hip/hip_runtime.h>

// ST-GCN block fused into two WMMA kernels for gfx1250 (wave32).
// Matrix math: v_wmma_f32_16x16x32_f16 (f16 operands, f32 accumulate).
// Intermediate z stored position-major [n][t][v][c] so kernel 2 stages its
// LDS tile with global_load_async_to_lds_b128 (no transpose, ASYNCcnt).

typedef __attribute__((ext_vector_type(16))) _Float16 v16h;
typedef __attribute__((ext_vector_type(8)))  _Float16 v8h;
typedef __attribute__((ext_vector_type(8)))  float    v8f;

#define EPSBN 1e-5f
#define N_  64
#define C_  64
#define T_  300
#define V_  25
#define K_  3
#define CO_ 64

// A-matrix 16x32 f16 from row-major [M][K] LDS (leading dim ld):
// lanes 0-15 hold rows 0-15 K={0..7,16..23}; lanes 16-31 K={8..15,24..31}.
// Contiguous 8-half runs -> ds_load_b128.
__device__ __forceinline__ v16h lda16(const _Float16* __restrict__ base, int ld, int lane) {
  const int r = lane & 15, hi = (lane >> 4) & 1;
  v16h a;
#pragma unroll
  for (int e = 0; e < 8; ++e) a[e]     = base[r * ld + hi * 8 + e];
#pragma unroll
  for (int e = 0; e < 8; ++e) a[e + 8] = base[r * ld + 16 + hi * 8 + e];
  return a;
}

// B-matrix 32x16 f16 from K-major [N][K] LDS (leading dim ld):
// lane (&15) = column; lanes 0-15 K={0..7,16..23}; lanes 16-31 K={8..15,24..31}.
__device__ __forceinline__ v16h ldb16t(const _Float16* __restrict__ base, int ld, int lane) {
  const int c = lane & 15, koff = ((lane >> 4) & 1) * 8;
  v16h b;
#pragma unroll
  for (int e = 0; e < 16; ++e) b[e] = base[c * ld + koff + (e < 8 ? e : e + 8)];
  return b;
}

// ---------------- Kernel 1: fused GCN (1x1 conv + graph agg + BN1 + ReLU) --
#define TT1  8
#define P1   200   // TT1*V_
#define P1P  208
#define XLD  72
#define WLD  72
#define YLD  264   // 8 t * 32 cols + 8 pad
#define ALD  40

__launch_bounds__(256)
__global__ void gcn_fused(const float* __restrict__ x, const float* __restrict__ A,
                          const float* __restrict__ cw, const float* __restrict__ cb,
                          const float* __restrict__ g1, const float* __restrict__ b1,
                          const float* __restrict__ m1, const float* __restrict__ v1,
                          _Float16* __restrict__ zr) {
  __shared__ _Float16 Xt[P1P * XLD];       // x tile transposed [p][c]
  __shared__ _Float16 Wl[192 * WLD];       // conv_w [o][c]
  __shared__ _Float16 Yl[192 * YLD];       // Y [o][t*32+v], v padded to 32
  __shared__ _Float16 Alt[K_ * 32 * ALD];  // A^T [k][w][v]
  __shared__ float    Asum[K_ * 32];       // sum_v A[k][v][w]

  const int n   = blockIdx.y;
  const int t0  = blockIdx.x * TT1;
  const int tid = threadIdx.x;
  const int lane = tid & 31, wave = tid >> 5;

  for (int i = tid; i < 192 * 64; i += 256) {
    int o = i >> 6, c = i & 63;
    Wl[o * WLD + c] = (_Float16)cw[i];
  }
  for (int i = tid; i < C_ * P1P; i += 256) {   // read x row-contiguous
    int c = i / P1P, p = i % P1P;
    float val = 0.f;
    if (p < P1) {
      int t = t0 + p / V_, v = p % V_;
      if (t < T_) val = x[((n * C_ + c) * T_ + t) * V_ + v];
    }
    Xt[p * XLD + c] = (_Float16)val;
  }
  for (int i = tid; i < K_ * 32 * ALD; i += 256) {
    int k = i / (32 * ALD), r = i % (32 * ALD), w = r / ALD, v = r % ALD;
    float val = (w < V_ && v < V_) ? A[((n * K_ + k) * V_ + v) * V_ + w] : 0.f;
    Alt[i] = (_Float16)val;
  }
  for (int i = tid; i < 192 * TT1 * 7; i += 256) {   // zero v-pad cols of Y
    int row = i / (TT1 * 7), r = i % (TT1 * 7), t = r / 7, v = 25 + r % 7;
    Yl[row * YLD + t * 32 + v] = (_Float16)0.f;
  }
  __syncthreads();

  if (tid < 96) {                                    // Asum[k][w]
    int k = tid >> 5, w = tid & 31;
    float s = 0.f;
#pragma unroll
    for (int v = 0; v < V_; ++v) s += (float)Alt[(k * 32 + w) * ALD + v];
    Asum[k * 32 + w] = s;
  }

  // phase 1: Y[o,p] = sum_c W[o,c] * X[c,p]
  for (int idx = wave; idx < 12 * 13; idx += 8) {
    int mt = idx % 12, nt = idx / 12;
    v8f acc = {};
#pragma unroll
    for (int kb = 0; kb < 2; ++kb) {
      v16h a = lda16(&Wl[(mt * 16) * WLD + kb * 32], WLD, lane);
      v16h b = ldb16t(&Xt[(nt * 16) * XLD + kb * 32], XLD, lane);
      acc = __builtin_amdgcn_wmma_f32_16x16x32_f16(false, a, false, b, (short)0,
                                                   acc, false, false);
    }
    const int hi = (lane >> 4) & 1;
    const int p  = nt * 16 + (lane & 15);
    if (p < P1) {
      const int col = (p / V_) * 32 + p % V_;
#pragma unroll
      for (int g = 0; g < 8; ++g)
        Yl[(mt * 16 + g + 8 * hi) * YLD + col] = (_Float16)acc[g];
    }
  }
  __syncthreads();

  // phase 2: Z[c,w] = sum_k Y_k[c,v] @ A_k[v,w]; one t per wave.
  const int tl = wave;
  const int t  = t0 + tl;
  if (t < T_) {
    const int hi = (lane >> 4) & 1, ln = lane & 15;
    v16h bv[K_][2];
#pragma unroll
    for (int k = 0; k < K_; ++k)
#pragma unroll
      for (int wt = 0; wt < 2; ++wt)
        bv[k][wt] = ldb16t(&Alt[(k * 32 + wt * 16) * ALD], ALD, lane);

#pragma unroll
    for (int ct = 0; ct < 4; ++ct) {
      v16h a0 = lda16(&Yl[(0 * 64 + ct * 16) * YLD + tl * 32], YLD, lane);
      v16h a1 = lda16(&Yl[(1 * 64 + ct * 16) * YLD + tl * 32], YLD, lane);
      v16h a2 = lda16(&Yl[(2 * 64 + ct * 16) * YLD + tl * 32], YLD, lane);
#pragma unroll
      for (int wt = 0; wt < 2; ++wt) {
        v8f acc = {};
        acc = __builtin_amdgcn_wmma_f32_16x16x32_f16(false, a0, false, bv[0][wt],
                                                     (short)0, acc, false, false);
        acc = __builtin_amdgcn_wmma_f32_16x16x32_f16(false, a1, false, bv[1][wt],
                                                     (short)0, acc, false, false);
        acc = __builtin_amdgcn_wmma_f32_16x16x32_f16(false, a2, false, bv[2][wt],
                                                     (short)0, acc, false, false);
        int w = wt * 16 + ln;
        if (w < V_) {
          v8h pk;                       // 8 consecutive channels -> one b128
#pragma unroll
          for (int g = 0; g < 8; ++g) {
            int c = ct * 16 + 8 * hi + g;
            float z = acc[g]
                    + cb[c]       * Asum[w]
                    + cb[64 + c]  * Asum[32 + w]
                    + cb[128 + c] * Asum[64 + w];
            float inv = g1[c] * rsqrtf(v1[c] + EPSBN);
            z = z * inv + (b1[c] - m1[c] * inv);
            pk[g] = (_Float16)fmaxf(z, 0.f);
          }
          // zr layout: [n][t][v][c]  (position-major, channels contiguous)
          *(v8h*)&zr[(((size_t)n * T_ + t) * V_ + w) * CO_ + ct * 16 + 8 * hi] = pk;
        }
      }
    }
  }
}

// ---------------- Kernel 2: fused TCN (9x1 conv + BN2 + residual + ReLU) --
#define TT2  16
#define ZROWS 600  // (16+8 halo t)*25 positions
#define ZLD  72    // row stride in halves (144 B, 16B-aligned, conflict-free)
#define WTLD 72

__launch_bounds__(256)
__global__ void tcn_fused(const _Float16* __restrict__ zr, const float* __restrict__ tw,
                          const float* __restrict__ tb,
                          const float* __restrict__ g2, const float* __restrict__ b2,
                          const float* __restrict__ m2, const float* __restrict__ v2,
                          const float* __restrict__ x, float* __restrict__ out) {
  __shared__ _Float16 Zt[ZROWS * ZLD];       // z tile+halo [p'][ci]
  __shared__ _Float16 Wt[9 * 64 * WTLD];     // taps [dt][co][ci]

  const int n   = blockIdx.y;
  const int t0  = blockIdx.x * TT2;
  const int tid = threadIdx.x;
  const int lane = tid & 31, wave = tid >> 5;

  // Stage z rows with async global->LDS 16B copies (zr is position-major, so
  // each LDS row is a straight copy of 64 contiguous channels = 8 x b128).
  {
    const _Float16* zb = zr + ((size_t)n * T_ + (t0 - 4)) * V_ * CO_;
    for (int s = tid; s < ZROWS * 8; s += 256) {
      int p = s >> 3, part = s & 7;            // row p', 16B chunk
      int t = t0 - 4 + p / V_;
      unsigned loff = (unsigned)(uintptr_t)&Zt[p * ZLD + part * 8];
      if (t >= 0 && t < T_) {
        const _Float16* g = zb + (size_t)p * CO_ + part * 8;
        asm volatile("global_load_async_to_lds_b128 %0, %1, off"
                     :: "v"(loff), "v"(g) : "memory");
      } else {
        *(v8h*)&Zt[p * ZLD + part * 8] = (v8h)0;  // zero halo out of range
      }
    }
  }
  for (int i = tid; i < 9 * 64 * 64; i += 256) {
    int dt = i / 4096, co = (i >> 6) & 63, ci = i & 63;
    Wt[(dt * 64 + co) * WTLD + ci] = (_Float16)tw[(co * 64 + ci) * 9 + dt];
  }
  asm volatile("s_wait_asynccnt 0x0" ::: "memory");
  __syncthreads();

  for (int idx = wave; idx < 100; idx += 8) {  // 4 co-tiles x 25 pos-tiles
    int mt = idx & 3, nt = idx >> 2;
    v8f acc = {};
    for (int dt = 0; dt < 9; ++dt) {
#pragma unroll
      for (int kb = 0; kb < 2; ++kb) {
        v16h a = lda16(&Wt[(dt * 64 + mt * 16) * WTLD + kb * 32], WTLD, lane);
        v16h b = ldb16t(&Zt[(nt * 16 + dt * V_) * ZLD + kb * 32], ZLD, lane);
        acc = __builtin_amdgcn_wmma_f32_16x16x32_f16(false, a, false, b, (short)0,
                                                     acc, false, false);
      }
    }
    const int hi = (lane >> 4) & 1, ln = lane & 15;
    int p = nt * 16 + ln, tt = p / V_, v = p % V_, t = t0 + tt;
    if (t < T_) {
#pragma unroll
      for (int g = 0; g < 8; ++g) {
        int c = mt * 16 + 8 * hi + g;
        float h = acc[g] + tb[c];
        float inv = g2[c] * rsqrtf(v2[c] + EPSBN);
        h = h * inv + (b2[c] - m2[c] * inv);
        h += x[((n * 64 + c) * T_ + t) * V_ + v];
        out[((n * 64 + c) * T_ + t) * V_ + v] = fmaxf(h, 0.f);
      }
    }
  }
}

extern "C" void kernel_launch(void* const* d_in, const int* in_sizes, int n_in,
                              void* d_out, int out_size, void* d_ws, size_t ws_size,
                              hipStream_t stream) {
  const float* x  = (const float*)d_in[0];
  const float* A  = (const float*)d_in[1];
  const float* cw = (const float*)d_in[2];
  const float* cb = (const float*)d_in[3];
  const float* g1 = (const float*)d_in[4];
  const float* b1 = (const float*)d_in[5];
  const float* m1 = (const float*)d_in[6];
  const float* v1 = (const float*)d_in[7];
  const float* tw = (const float*)d_in[8];
  const float* tb = (const float*)d_in[9];
  const float* g2 = (const float*)d_in[10];
  const float* b2 = (const float*)d_in[11];
  const float* m2 = (const float*)d_in[12];
  const float* v2 = (const float*)d_in[13];
  float* out = (float*)d_out;
  _Float16* zr = (_Float16*)d_ws;   // [n][t][v][c] f16 = 61.44 MB

  gcn_fused<<<dim3((T_ + TT1 - 1) / TT1, N_), dim3(256), 0, stream>>>(
      x, A, cw, cb, g1, b1, m1, v1, zr);
  tcn_fused<<<dim3((T_ + TT2 - 1) / TT2, N_), dim3(256), 0, stream>>>(
      zr, tw, tb, g2, b2, m2, v2, x, out);

  hipMemcpyAsync(out + (size_t)N_ * CO_ * T_ * V_, A,
                 (size_t)N_ * K_ * V_ * V_ * sizeof(float),
                 hipMemcpyDeviceToDevice, stream);
}
